// EncoderImagePrecompAttn_44530220925148
// MI455X (gfx1250) — matrix operand order repaired and checked by reading, not compiled, not run on hardware
//
#include <hip/hip_runtime.h>
#include <hip/hip_bf16.h>

typedef unsigned short u16t;
typedef unsigned int   u32t;
typedef __attribute__((ext_vector_type(16))) __bf16 v16bf;
typedef __attribute__((ext_vector_type(8)))  float  v8f;
typedef int v4i __attribute__((vector_size(16)));   // matches builtin param type

#define EMBED 2048
#define E3    6144
#define BATCH 256
#define LREG  36
#define BLR   (BATCH*LREG)   // 9216
#define ASPACE 500
#define INV_TEMP 1.25f
#define BN_EPS 1e-5f

// CDNA5 async global->LDS path (ASYNCcnt), with safe fallback
#if __has_builtin(__builtin_amdgcn_global_load_async_to_lds_b128) && \
    __has_builtin(__builtin_amdgcn_s_wait_asynccnt)
#define USE_ASYNC_LDS 1
#else
#define USE_ASYNC_LDS 0
#endif

#define GP1 __attribute__((address_space(1))) v4i*
#define LP3 __attribute__((address_space(3))) v4i*

// ---------------- helpers ----------------
__device__ __forceinline__ u16t f2bf(float f) {
    union { float f; u32t u; } v; v.f = f;
    u32t u = v.u;
    u32t r = (u + 0x7FFFu + ((u >> 16) & 1u)) >> 16;   // RNE
    return (u16t)r;
}
__device__ __forceinline__ float sigmoidf(float x) { return 1.0f / (1.0f + __expf(-x)); }

// ---------------- f32 -> bf16 convert ----------------
__global__ void k_cvt_bf16(const float* __restrict__ s, u16t* __restrict__ d, int n) {
    int i = blockIdx.x * blockDim.x + threadIdx.x;
    int stride = gridDim.x * blockDim.x;
    for (; i < n; i += stride) d[i] = f2bf(s[i]);
}

// slice step t of [B,L,E] f32 into [B,E] bf16
__global__ void k_slice_cvt(const float* __restrict__ x, u16t* __restrict__ d,
                            int t, int B, int L, int E) {
    int i = blockIdx.x * blockDim.x + threadIdx.x;
    int n = B * E;
    int stride = gridDim.x * blockDim.x;
    for (; i < n; i += stride) {
        int b = i / E, c = i - b * E;
        d[i] = f2bf(x[((size_t)b * L + t) * (size_t)E + c]);
    }
}

// ---------------- WMMA bf16 GEMM:  C[M,N] = A[M,K] * W[N,K]^T + bias ----------------
#define TM 128
#define TN 128
#define TK 32

__global__ void __launch_bounds__(256)
k_gemm_bf16(const u16t* __restrict__ A, const u16t* __restrict__ W,
            const float* __restrict__ bias, float* __restrict__ C,
            int M, int N, int K) {
#if USE_ASYNC_LDS
    __shared__ u16t sA[2][TM * TK];
    __shared__ u16t sB[2][TN * TK];
#else
    __shared__ u16t sA[1][TM * TK];
    __shared__ u16t sB[1][TN * TK];
#endif

    const int t    = threadIdx.x;
    const int lane = t & 31;
    const int wave = t >> 5;        // 0..7
    const int wm   = wave >> 1;     // 0..3 (32-row slabs)
    const int wn   = wave & 1;      // 0..1 (64-col slabs)
    const int mBase = blockIdx.y * TM;
    const int nBase = blockIdx.x * TN;

    v8f acc[2][4] = {};

    const int lr = t >> 1;              // 0..127 tile row
    const int lk = (t & 1) * 16;        // 0 or 16 (halfs)

    // Clamp edge rows instead of zero-filling: padding rows/cols only feed
    // accumulators whose results are never stored, so the stored region is
    // unaffected and the load path stays branch/divergence free.
    const int rowA = (mBase + lr < M) ? (mBase + lr) : (M - 1);
    const int rowW = (nBase + lr < N) ? (nBase + lr) : (N - 1);
    const u16t* gA = A + (size_t)rowA * K + lk;
    const u16t* gW = W + (size_t)rowW * K + lk;

#if USE_ASYNC_LDS
    // Software-pipelined async staging: issue tile k+1, retire tile k
    // (4 async instructions in flight per wave per tile -> wait cnt<=4).
    {
        u16t* la = &sA[0][lr * TK + lk];
        u16t* lb = &sB[0][lr * TK + lk];
        __builtin_amdgcn_global_load_async_to_lds_b128((GP1)(gA), (LP3)la, 0, 0);
        __builtin_amdgcn_global_load_async_to_lds_b128((GP1)(gA), (LP3)la, 16, 0);
        __builtin_amdgcn_global_load_async_to_lds_b128((GP1)(gW), (LP3)lb, 0, 0);
        __builtin_amdgcn_global_load_async_to_lds_b128((GP1)(gW), (LP3)lb, 16, 0);
    }
#endif

    for (int kk = 0; kk < K; kk += TK) {
#if USE_ASYNC_LDS
        const int buf = (kk / TK) & 1;
        if (kk + TK < K) {
            const int nb = buf ^ 1;
            u16t* la = &sA[nb][lr * TK + lk];
            u16t* lb = &sB[nb][lr * TK + lk];
            __builtin_amdgcn_global_load_async_to_lds_b128((GP1)(gA + kk + TK), (LP3)la, 0, 0);
            __builtin_amdgcn_global_load_async_to_lds_b128((GP1)(gA + kk + TK), (LP3)la, 16, 0);
            __builtin_amdgcn_global_load_async_to_lds_b128((GP1)(gW + kk + TK), (LP3)lb, 0, 0);
            __builtin_amdgcn_global_load_async_to_lds_b128((GP1)(gW + kk + TK), (LP3)lb, 16, 0);
            __builtin_amdgcn_s_wait_asynccnt(4);   // current tile retired
        } else {
            __builtin_amdgcn_s_wait_asynccnt(0);   // drain
        }
        __syncthreads();
#else
        const int buf = 0;
        {
            const uint4* srcA = (const uint4*)(gA + kk);
            uint4 a0 = srcA[0], a1 = srcA[1];
            const uint4* srcW = (const uint4*)(gW + kk);
            uint4 w0 = srcW[0], w1 = srcW[1];
            if (kk + TK < K) {
                __builtin_prefetch((const void*)(gA + kk + TK), 0, 1);
                __builtin_prefetch((const void*)(gW + kk + TK), 0, 1);
            }
            uint4* da = (uint4*)(&sA[0][lr * TK + lk]);
            da[0] = a0; da[1] = a1;
            uint4* db = (uint4*)(&sB[0][lr * TK + lk]);
            db[0] = w0; db[1] = w1;
        }
        __syncthreads();
#endif

        // ---- build fragments per 16-bit A(16x32)/B(32x16) lane layouts ----
        union FragU { uint4 q[2]; v16bf v; };
        const int mrow = lane & 15;
        const int hA   = (lane >> 4) << 3;   // 0 or 8  (K half-group)
        const int kB   = (lane >> 4) << 3;   // 0 or 8  (K half-group for B)

        FragU a[2], b[4];
        #pragma unroll
        for (int i = 0; i < 2; i++) {
            const u16t* p = &sA[buf][(wm * 32 + i * 16 + mrow) * TK];
            a[i].q[0] = *(const uint4*)(p + hA);
            a[i].q[1] = *(const uint4*)(p + hA + 16);
        }
        #pragma unroll
        for (int j = 0; j < 4; j++) {
            const u16t* p = &sB[buf][(wn * 64 + j * 16 + mrow) * TK + kB];
            b[j].q[0] = *(const uint4*)(p);
            b[j].q[1] = *(const uint4*)(p + 16);
        }

        #pragma unroll
        for (int i = 0; i < 2; i++)
            #pragma unroll
            for (int j = 0; j < 4; j++)
                acc[i][j] = __builtin_amdgcn_wmma_f32_16x16x32_bf16(
                    false, a[i].v, false, b[j].v, (short)0, acc[i][j], false, false);
        __syncthreads();
    }

    // ---- epilogue: C/D layout VGPR r -> M = r + 8*(lane>=16), N = lane&15 ----
    const int rOff = (lane >> 4) << 3;
    const int col  = lane & 15;
    #pragma unroll
    for (int i = 0; i < 2; i++) {
        #pragma unroll
        for (int j = 0; j < 4; j++) {
            int n = nBase + wn * 64 + j * 16 + col;
            if (n >= N) continue;
            float bv = bias ? bias[n] : 0.0f;
            #pragma unroll
            for (int r = 0; r < 8; r++) {
                int m = mBase + wm * 32 + i * 16 + rOff + r;
                if (m < M) C[(size_t)m * N + n] = acc[i][j][r] + bv;
            }
        }
    }
}

// ---------------- l2 norm over region axis (dim 1 of [B,L,C]) ----------------
__global__ void k_l2norm(float* __restrict__ x, float* __restrict__ out2,
                         int B, int L, int C) {
    int i = blockIdx.x * blockDim.x + threadIdx.x;
    int n = B * C;
    int stride = gridDim.x * blockDim.x;
    for (; i < n; i += stride) {
        int b = i / C, c = i - b * C;
        size_t base = (size_t)b * L * C + c;
        float s = 0.0f;
        for (int l = 0; l < L; l++) { float v = x[base + (size_t)l * C]; s += v * v; }
        float inv = rsqrtf(s);
        for (int l = 0; l < L; l++) {
            size_t idx = base + (size_t)l * C;
            float v = x[idx] * inv;
            x[idx] = v;
            if (out2) out2[idx] = v;
        }
    }
}

// ---------------- affinity R[b,l,m] = dot(th[b,l,:], ph[b,m,:]) / L ----------------
__global__ void __launch_bounds__(288)
k_affinity(const float* __restrict__ th, const float* __restrict__ ph,
           float* __restrict__ R, int B, int L, int C) {
    int bl = blockIdx.x;
    int b = bl / L, l = bl - b * L;
    int t = threadIdx.x;
    int m = t >> 3, sub = t & 7;
    __shared__ float red[288];
    float acc = 0.0f;
    if (m < L) {
        const float* tr = th + ((size_t)b * L + l) * C;
        const float* pr = ph + ((size_t)b * L + m) * C;
        for (int k = sub; k < C; k += 8) acc += tr[k] * pr[k];
    }
    red[t] = acc;
    __syncthreads();
    if (sub == 0 && m < L) {
        float s = 0.0f;
        for (int u = 0; u < 8; u++) s += red[(m << 3) + u];
        R[((size_t)b * L + l) * L + m] = s / (float)L;
    }
}

// ---------------- y[b,l,c] = sum_m R[b,l,m] * gv[b,m,c] ----------------
__global__ void k_attend(const float* __restrict__ R, const float* __restrict__ gv,
                         float* __restrict__ y, int B, int L, int C) {
    size_t n = (size_t)B * L * C;
    size_t i = (size_t)blockIdx.x * blockDim.x + threadIdx.x;
    size_t stride = (size_t)gridDim.x * blockDim.x;
    for (; i < n; i += stride) {
        int c = (int)(i % C);
        int l = (int)((i / C) % L);
        int b = (int)(i / ((size_t)C * L));
        const float* Rr = R + ((size_t)b * L + l) * L;
        const float* g0 = gv + (size_t)b * L * C + c;
        float s = 0.0f;
        for (int m = 0; m < L; m++) s += Rr[m] * g0[(size_t)m * C];
        y[i] = s;
    }
}

// ---------------- per-column mean/var over rows ----------------
__global__ void __launch_bounds__(256)
k_colstats(const float* __restrict__ X, float* __restrict__ mean,
           float* __restrict__ var, int rows, int C) {
    int c = blockIdx.x;
    int t = threadIdx.x;
    __shared__ float s1[256], s2[256];
    float a = 0.0f, b = 0.0f;
    for (int r = t; r < rows; r += 256) {
        float v = X[(size_t)r * C + c];
        a += v; b += v * v;
    }
    s1[t] = a; s2[t] = b;
    __syncthreads();
    for (int o = 128; o > 0; o >>= 1) {
        if (t < o) { s1[t] += s1[t + o]; s2[t] += s2[t + o]; }
        __syncthreads();
    }
    if (t == 0) {
        float m = s1[0] / (float)rows;
        mean[c] = m;
        var[c]  = s2[0] / (float)rows - m * m;
    }
}

// ---------------- BN apply (+ optional residual) ----------------
__global__ void k_bn_apply(const float* __restrict__ X, const float* __restrict__ mean,
                           const float* __restrict__ var, const float* __restrict__ g,
                           const float* __restrict__ bta, const float* __restrict__ resid,
                           float* __restrict__ out, int rows, int C) {
    size_t n = (size_t)rows * C;
    size_t i = (size_t)blockIdx.x * blockDim.x + threadIdx.x;
    size_t stride = (size_t)gridDim.x * blockDim.x;
    for (; i < n; i += stride) {
        int c = (int)(i % C);
        float v = (X[i] - mean[c]) * rsqrtf(var[c] + BN_EPS) * g[c] + bta[c];
        if (resid) v += resid[i];
        out[i] = v;
    }
}

// ---------------- GRU gate fusion (gi/gh include biases) ----------------
__global__ void k_gru(const float* __restrict__ gi, const float* __restrict__ gh,
                      float* __restrict__ h, int B, int E) {
    int i = blockIdx.x * blockDim.x + threadIdx.x;
    int n = B * E;
    int stride = gridDim.x * blockDim.x;
    for (; i < n; i += stride) {
        int b = i / E, c = i - b * E;
        const float* gib = gi + (size_t)b * 3 * E;
        const float* ghb = gh + (size_t)b * 3 * E;
        float r = sigmoidf(gib[c] + ghb[c]);
        float z = sigmoidf(gib[E + c] + ghb[E + c]);
        float nn = tanhf(gib[2 * E + c] + r * ghb[2 * E + c]);
        h[i] = (1.0f - z) * nn + z * h[i];
    }
}

// ---------------- feat = relu(concat(img_t, h)) -> bf16 ----------------
__global__ void k_featcat(const float* __restrict__ gcn, const float* __restrict__ h,
                          u16t* __restrict__ featbf, int t, int B, int L, int E) {
    int i = blockIdx.x * blockDim.x + threadIdx.x;
    int E2 = 2 * E;
    int n = B * E2;
    int stride = gridDim.x * blockDim.x;
    for (; i < n; i += stride) {
        int b = i / E2, c = i - b * E2;
        float v = (c < E) ? gcn[((size_t)b * L + t) * E + c]
                          : h[(size_t)b * E + (c - E)];
        v = v > 0.0f ? v : 0.0f;
        featbf[i] = f2bf(v);
    }
}

// ---------------- softmax/argmax head: action, logp, attention weight ----------------
__global__ void __launch_bounds__(128)
k_actions(const float* __restrict__ logits, float* __restrict__ att,
          float* __restrict__ out_logp, float* __restrict__ out_act,
          int t, int B, int L) {
    int bh = blockIdx.x;
    int b = bh >> 1, h = bh & 1;
    const float* lg = logits + (size_t)b * (2 * ASPACE) + h * ASPACE;
    int tid = threadIdx.x;
    __shared__ float smax[128]; __shared__ int sidx[128]; __shared__ float ssum[128];
    float mx = -1e30f; int mi = 0;
    for (int i = tid; i < ASPACE; i += 128) {
        float v = lg[i] * INV_TEMP;
        if (v > mx) { mx = v; mi = i; }
    }
    smax[tid] = mx; sidx[tid] = mi;
    __syncthreads();
    for (int o = 64; o > 0; o >>= 1) {
        if (tid < o && smax[tid + o] > smax[tid]) { smax[tid] = smax[tid + o]; sidx[tid] = sidx[tid + o]; }
        __syncthreads();
    }
    float gmax = smax[0]; int gidx = sidx[0];
    float s = 0.0f;
    for (int i = tid; i < ASPACE; i += 128) s += __expf(lg[i] * INV_TEMP - gmax);
    ssum[tid] = s;
    __syncthreads();
    for (int o = 64; o > 0; o >>= 1) {
        if (tid < o) ssum[tid] += ssum[tid + o];
        __syncthreads();
    }
    if (tid == 0) {
        float sum = ssum[0];
        float logp = -__logf(sum);                      // log(max softmax prob)
        float action = (float)gidx / (float)ASPACE;     // straight-through argmax
        att[(size_t)b * 2 + h] = 20.0f * sigmoidf(action);
        size_t o = (size_t)b * (2 * L) + (size_t)t * 2 + h; // reshape(B,HEAD,L) memory order
        out_logp[o] = logp;
        out_act[o]  = action;
    }
}

// ---------------- img_w[t] = img_t * att per head-half ----------------
__global__ void k_scale_img(const float* __restrict__ gcn, const float* __restrict__ att,
                            float* __restrict__ imgw, int t, int B, int L, int E) {
    int i = blockIdx.x * blockDim.x + threadIdx.x;
    int n = B * E;
    int stride = gridDim.x * blockDim.x;
    for (; i < n; i += stride) {
        int b = i / E, c = i - b * E;
        float a = att[(size_t)b * 2 + (c >= (E >> 1) ? 1 : 0)];
        imgw[(size_t)t * B * E + i] = gcn[((size_t)b * L + t) * E + c] * a;
    }
}

// ---------------- feats = mean_t imgw[t] + h2 ----------------
__global__ void k_mean_add(const float* __restrict__ imgw, const float* __restrict__ h2,
                           float* __restrict__ feats, int Ls, int B, int E) {
    int i = blockIdx.x * blockDim.x + threadIdx.x;
    int n = B * E;
    int stride = gridDim.x * blockDim.x;
    for (; i < n; i += stride) {
        float s = 0.0f;
        for (int t = 0; t < Ls; t++) s += imgw[(size_t)t * B * E + i];
        feats[i] = s / (float)Ls + h2[i];
    }
}

__global__ void k_fill(float* __restrict__ p, float v, int n) {
    int i = blockIdx.x * blockDim.x + threadIdx.x;
    int stride = gridDim.x * blockDim.x;
    for (; i < n; i += stride) p[i] = v;
}

// ================== host orchestration ==================
static inline int gs(size_t n, int block) {
    size_t g = (n + block - 1) / block;
    return (int)(g > 65535u * 16u ? 65535u * 16u : g);
}

extern "C" void kernel_launch(void* const* d_in, const int* in_sizes, int n_in,
                              void* d_out, int out_size, void* d_ws, size_t ws_size,
                              hipStream_t stream) {
    const int B = BATCH, L = LREG, E = EMBED;
    const float* images = (const float*)d_in[0];
    const float* fc_w = (const float*)d_in[1];
    const float* fc_b = (const float*)d_in[2];
    const float *g_w[4], *g_b[4], *th_w[4], *th_b[4], *ph_w[4], *ph_b[4],
                *w_w[4], *w_b[4], *bng[4], *bnb[4];
    for (int i = 0; i < 4; i++) {
        int base = 3 + i * 10;
        g_w[i]  = (const float*)d_in[base + 0]; g_b[i]  = (const float*)d_in[base + 1];
        th_w[i] = (const float*)d_in[base + 2]; th_b[i] = (const float*)d_in[base + 3];
        ph_w[i] = (const float*)d_in[base + 4]; ph_b[i] = (const float*)d_in[base + 5];
        w_w[i]  = (const float*)d_in[base + 6]; w_b[i]  = (const float*)d_in[base + 7];
        bng[i]  = (const float*)d_in[base + 8]; bnb[i]  = (const float*)d_in[base + 9];
    }
    const float* cell_wih = (const float*)d_in[43];
    const float* cell_bih = (const float*)d_in[44];
    const float* cell_whh = (const float*)d_in[45];
    const float* cell_bhh = (const float*)d_in[46];
    const float* mus_w    = (const float*)d_in[47];
    const float* mus_b    = (const float*)d_in[48];
    const float* rnn_wih  = (const float*)d_in[49];
    const float* rnn_bih  = (const float*)d_in[50];
    const float* rnn_whh  = (const float*)d_in[51];
    const float* rnn_bhh  = (const float*)d_in[52];
    const float* fbn_g    = (const float*)d_in[53];
    const float* fbn_b    = (const float*)d_in[54];

    // ---- workspace carve-out ----
    char* wp = (char*)d_ws;
    auto carve = [&](size_t bytes) -> void* {
        void* p = (void*)wp;
        wp += (bytes + 255) & ~(size_t)255;
        return p;
    };
    u16t* fcw_bf = (u16t*)carve((size_t)E * E * 2);
    u16t *gw_bf[4], *thw_bf[4], *phw_bf[4], *ww_bf[4];
    for (int i = 0; i < 4; i++) {
        gw_bf[i]  = (u16t*)carve((size_t)E * E * 2);
        thw_bf[i] = (u16t*)carve((size_t)E * E * 2);
        phw_bf[i] = (u16t*)carve((size_t)E * E * 2);
        ww_bf[i]  = (u16t*)carve((size_t)E * E * 2);
    }
    u16t* cwih_bf = (u16t*)carve((size_t)E3 * E * 2);
    u16t* cwhh_bf = (u16t*)carve((size_t)E3 * E * 2);
    u16t* rwih_bf = (u16t*)carve((size_t)E3 * E * 2);
    u16t* rwhh_bf = (u16t*)carve((size_t)E3 * E * 2);
    u16t* musw_bf = (u16t*)carve((size_t)(2 * ASPACE) * (2 * E) * 2);

    u16t* xbf   = (u16t*)carve((size_t)BLR * E * 2);
    float* xf   = (float*)carve((size_t)BLR * E * 4);
    float* gv   = (float*)carve((size_t)BLR * E * 4);
    float* thb  = (float*)carve((size_t)BLR * E * 4);
    float* phb  = (float*)carve((size_t)BLR * E * 4);
    float* wy   = (float*)carve((size_t)BLR * E * 4);
    float* Rbuf = (float*)carve((size_t)B * L * L * 4);
    float* meanb = (float*)carve((size_t)E * 4);
    float* varb  = (float*)carve((size_t)E * 4);
    float* h1   = (float*)carve((size_t)B * E * 4);
    float* h2   = (float*)carve((size_t)B * E * 4);
    u16t* hbf   = (u16t*)carve((size_t)B * E * 2);
    float* gi   = (float*)carve((size_t)B * E3 * 4);
    float* gh   = (float*)carve((size_t)B * E3 * 4);
    u16t* featbf = (u16t*)carve((size_t)B * 2 * E * 2);
    float* logits = (float*)carve((size_t)B * 2 * ASPACE * 4);
    float* attb   = (float*)carve((size_t)B * 2 * 4);
    float* imgw   = (float*)carve((size_t)L * B * E * 4);
    float* feats  = (float*)carve((size_t)B * E * 4);
    (void)ws_size; (void)n_in; (void)in_sizes; (void)out_size;

    float* out_feats = (float*)d_out;                              // [256,2048]
    float* out_gcn   = out_feats + (size_t)B * E;                  // [256,36,2048]
    float* out_logp  = out_gcn + (size_t)B * L * E;                // [256,2,36]
    float* out_act   = out_logp + (size_t)B * 2 * L;               // [256,2,36]

    const int blk = 256;
    auto cvt = [&](const float* s, u16t* d, size_t n) {
        k_cvt_bf16<<<gs(n, blk), blk, 0, stream>>>(s, d, (int)n);
    };
    auto gemm = [&](const u16t* A, const u16t* W, const float* bias, float* C,
                    int M, int N, int K) {
        dim3 g((N + TN - 1) / TN, (M + TM - 1) / TM);
        k_gemm_bf16<<<g, 256, 0, stream>>>(A, W, bias, C, M, N, K);
    };

    // ---- weight conversion (bf16) ----
    cvt(fc_w, fcw_bf, (size_t)E * E);
    for (int i = 0; i < 4; i++) {
        cvt(g_w[i], gw_bf[i], (size_t)E * E);
        cvt(th_w[i], thw_bf[i], (size_t)E * E);
        cvt(ph_w[i], phw_bf[i], (size_t)E * E);
        cvt(w_w[i], ww_bf[i], (size_t)E * E);
    }
    cvt(cell_wih, cwih_bf, (size_t)E3 * E);
    cvt(cell_whh, cwhh_bf, (size_t)E3 * E);
    cvt(rnn_wih, rwih_bf, (size_t)E3 * E);
    cvt(rnn_whh, rwhh_bf, (size_t)E3 * E);
    cvt(mus_w, musw_bf, (size_t)(2 * ASPACE) * (2 * E));

    // ---- FC + l2norm ----
    cvt(images, xbf, (size_t)BLR * E);
    gemm(xbf, fcw_bf, fc_b, xf, BLR, E, E);
    k_l2norm<<<gs((size_t)B * E, blk), blk, 0, stream>>>(xf, (float*)nullptr, B, L, E);

    // ---- 4 GCN non-local blocks ----
    for (int i = 0; i < 4; i++) {
        cvt(xf, xbf, (size_t)BLR * E);
        gemm(xbf, gw_bf[i], g_b[i], gv, BLR, E, E);
        gemm(xbf, thw_bf[i], th_b[i], thb, BLR, E, E);
        gemm(xbf, phw_bf[i], ph_b[i], phb, BLR, E, E);
        k_affinity<<<B * L, 288, 0, stream>>>(thb, phb, Rbuf, B, L, E);
        k_attend<<<gs((size_t)BLR * E, blk), blk, 0, stream>>>(Rbuf, gv, thb, B, L, E);
        cvt(thb, xbf, (size_t)BLR * E);
        gemm(xbf, ww_bf[i], w_b[i], wy, BLR, E, E);
        k_colstats<<<E, 256, 0, stream>>>(wy, meanb, varb, BLR, E);
        k_bn_apply<<<gs((size_t)BLR * E, blk), blk, 0, stream>>>(
            wy, meanb, varb, bng[i], bnb[i], xf, xf, BLR, E);
    }
    // gcn_emd = l2norm(x), also copied into output
    k_l2norm<<<gs((size_t)B * E, blk), blk, 0, stream>>>(xf, out_gcn, B, L, E);

    // ---- attention GRU scan ----
    k_fill<<<gs((size_t)B * E, blk), blk, 0, stream>>>(h1, 0.0f, B * E);
    k_fill<<<gs((size_t)B * E, blk), blk, 0, stream>>>(h2, 0.0f, B * E);
    for (int t = 0; t < L; t++) {
        k_slice_cvt<<<gs((size_t)B * E, blk), blk, 0, stream>>>(xf, xbf, t, B, L, E);
        gemm(xbf, cwih_bf, cell_bih, gi, B, E3, E);
        cvt(h1, hbf, (size_t)B * E);
        gemm(hbf, cwhh_bf, cell_bhh, gh, B, E3, E);
        k_gru<<<gs((size_t)B * E, blk), blk, 0, stream>>>(gi, gh, h1, B, E);
        k_featcat<<<gs((size_t)B * 2 * E, blk), blk, 0, stream>>>(xf, h1, featbf, t, B, L, E);
        gemm(featbf, musw_bf, mus_b, logits, B, 2 * ASPACE, 2 * E);
        k_actions<<<B * 2, 128, 0, stream>>>(logits, attb, out_logp, out_act, t, B, L);
        k_scale_img<<<gs((size_t)B * E, blk), blk, 0, stream>>>(xf, attb, imgw, t, B, L, E);
    }

    // ---- second GRU over weighted image sequence ----
    for (int t = 0; t < L; t++) {
        cvt(imgw + (size_t)t * B * E, xbf, (size_t)B * E);
        gemm(xbf, rwih_bf, rnn_bih, gi, B, E3, E);
        cvt(h2, hbf, (size_t)B * E);
        gemm(hbf, rwhh_bf, rnn_bhh, gh, B, E3, E);
        k_gru<<<gs((size_t)B * E, blk), blk, 0, stream>>>(gi, gh, h2, B, E);
    }

    // ---- feats = mean(img_seq) + h_last, then batchnorm over batch ----
    k_mean_add<<<gs((size_t)B * E, blk), blk, 0, stream>>>(imgw, h2, feats, L, B, E);
    k_colstats<<<E, 256, 0, stream>>>(feats, meanb, varb, B, E);
    k_bn_apply<<<gs((size_t)B * E, blk), blk, 0, stream>>>(
        feats, meanb, varb, fbn_g, fbn_b, (const float*)nullptr, out_feats, B, E);
}
